// LatentEncoder_27728308863585
// MI455X (gfx1250) — matrix operand order repaired and checked by reading, not compiled, not run on hardware
//
#include <hip/hip_runtime.h>
#include <hip/hip_bf16.h>
#include <math.h>

typedef __attribute__((ext_vector_type(16))) __bf16 v16bf;
typedef __attribute__((ext_vector_type(8)))  float  v8f;

#define N_ROWS 4096
#define NOK 8
#define NUM_CLASSES 20
#define EPSF 1e-7f

// ---------------------------------------------------------------------------
// WMMA GEMM: C[M x N] = A[M x K] @ W[N x K]^T  (+ epilogue)
// One wave (32 threads) computes a 32x64 tile:
//   2 M-subtiles x 4 N-subtiles = 8 WMMA accumulators; each B fragment is
//   reused across both M-subtiles (halves B traffic vs 16x64 tiling).
// epilogue: 1 = relu(C + bias);  2 = C + bias + deltas@Wdel^T + bdel + scale*Wsc + bsc
// ---------------------------------------------------------------------------
__global__ __launch_bounds__(32)
void wmma_gemm_kernel(const float* __restrict__ A, int lda,
                      const float* __restrict__ W,
                      float* __restrict__ C, int ldc,
                      int K,
                      const float* __restrict__ bias,
                      int epilogue,
                      const float* __restrict__ deltas,
                      const float* __restrict__ Wdel,
                      const float* __restrict__ bdel,
                      const float* __restrict__ scale,
                      const float* __restrict__ Wsc,
                      const float* __restrict__ bsc)
{
    const int lane = threadIdx.x;      // 0..31
    const int g    = lane >> 4;        // lane group (0/1)
    const int l    = lane & 15;
    const int m0   = blockIdx.y * 32;
    const int n0   = blockIdx.x * 64;

    v8f acc[2][4];
#pragma unroll
    for (int u = 0; u < 2; ++u)
#pragma unroll
        for (int t = 0; t < 4; ++t) acc[u][t] = v8f{};

    const float* Arow0 = A + (size_t)(m0 + l) * lda;
    const float* Arow1 = A + (size_t)(m0 + 16 + l) * lda;

    for (int k0 = 0; k0 < K; k0 += 32) {
        // A fragments, 16-bit A 16x32 layout:
        // lanes 0-15: v0..3 -> K=0..7, v4..7 -> K=16..23 ; lanes 16-31: +8
        v16bf a0, a1;
#pragma unroll
        for (int v = 0; v < 8; ++v) {
            const int kk = k0 + 16 * (v >> 2) + 8 * g + 2 * (v & 3);
            a0[2 * v]     = (__bf16)Arow0[kk];
            a0[2 * v + 1] = (__bf16)Arow0[kk + 1];
            a1[2 * v]     = (__bf16)Arow1[kk];
            a1[2 * v + 1] = (__bf16)Arow1[kk + 1];
        }
        // B fragment: B[k][n] = W[n][k]; same K layout as A: e0..7 -> K=8g+e, e8..15 -> K=16+8g+(e-8)
#pragma unroll
        for (int t = 0; t < 4; ++t) {
            const float* Wrow = W + (size_t)(n0 + 16 * t + l) * K + k0;
            v16bf b;
#pragma unroll
            for (int e = 0; e < 16; ++e) b[e] = (__bf16)Wrow[8 * g + (e & 7) + 16 * (e >> 3)];
            acc[0][t] = __builtin_amdgcn_wmma_f32_16x16x32_bf16(
                false, a0, false, b, (short)0, acc[0][t], false, false);
            acc[1][t] = __builtin_amdgcn_wmma_f32_16x16x32_bf16(
                false, a1, false, b, (short)0, acc[1][t], false, false);
        }
    }

    // C/D layout: element r of lane -> row = m0 + 16*u + 8*g + r, col = n0 + 16*t + l
#pragma unroll
    for (int u = 0; u < 2; ++u) {
#pragma unroll
        for (int t = 0; t < 4; ++t) {
            const int col = n0 + 16 * t + l;
#pragma unroll
            for (int r = 0; r < 8; ++r) {
                const int row = m0 + 16 * u + 8 * g + r;
                float val = acc[u][t][r];
                if (epilogue == 1) {
                    val = fmaxf(val + bias[col], 0.0f);
                } else { // epilogue == 2
                    val += bias[col];
                    const float* dm = deltas + (size_t)row * 4;
                    const float* wd = Wdel + (size_t)col * 4;
                    val += dm[0] * wd[0] + dm[1] * wd[1] + dm[2] * wd[2] + dm[3] * wd[3];
                    val += bdel[col];
                    val += scale[row] * Wsc[col] + bsc[col];
                }
                C[(size_t)row * ldc + col] = val;
            }
        }
    }
}

// ---------------------------------------------------------------------------
// Attention layer 2 + softmax(logits/temp): alphas[M x 8]
// ---------------------------------------------------------------------------
__global__ __launch_bounds__(256)
void attn_l2_softmax_kernel(const float* __restrict__ Hm, int Hdim,
                            const float* __restrict__ W2,   // [8 x Hdim]
                            const float* __restrict__ b2,   // [8]
                            const int* __restrict__ iter_ptr,
                            float* __restrict__ alphas)     // [M x 8]
{
    extern __shared__ float sW[];
    for (int i = threadIdx.x; i < NOK * Hdim; i += 256) sW[i] = W2[i];
    __syncthreads();

    const int m = blockIdx.x * 256 + threadIdx.x;
    const float* hrow = Hm + (size_t)m * Hdim;
    float acc[NOK] = {0.f, 0.f, 0.f, 0.f, 0.f, 0.f, 0.f, 0.f};
    for (int c = 0; c < Hdim; ++c) {
        const float hv = hrow[c];
#pragma unroll
        for (int j = 0; j < NOK; ++j) acc[j] += hv * sW[j * Hdim + c];
    }
    const float it   = (float)(*iter_ptr);
    const float temp = fmaxf(1.0f, 30.0f * (90000.0f - it) * (1.0f / 90000.0f));
    const float invt = 1.0f / temp;
    float mx = -1e30f;
#pragma unroll
    for (int j = 0; j < NOK; ++j) {
        acc[j] = (acc[j] + b2[j]) * invt;
        mx = fmaxf(mx, acc[j]);
    }
    float s = 0.f;
#pragma unroll
    for (int j = 0; j < NOK; ++j) { acc[j] = __expf(acc[j] - mx); s += acc[j]; }
    const float inv = 1.0f / s;
#pragma unroll
    for (int j = 0; j < NOK; ++j) alphas[(size_t)m * NOK + j] = acc[j] * inv;
}

// ---------------------------------------------------------------------------
// Contrastive loss, decomposed (x = P P^T rank-8; gt rank-4 or class-equality).
// Pass 1: column sums of exp(2x) and exp(2gt).
// Pass 2: S = sum_ij (2 x_ij - log denomx_j) * exp(2 gt_ij) / denomg_j
// loss = -S / N
// ---------------------------------------------------------------------------
template <bool CLS>
__global__ __launch_bounds__(256)
void loss_pass1_kernel(const float* __restrict__ P,
                       const int* __restrict__ cls,
                       const float* __restrict__ D,
                       float* __restrict__ colsum_x,
                       float* __restrict__ colsum_g)
{
    __shared__ float sP[256 * NOK];
    __shared__ float sD[256 * 4];
    __shared__ int   sC[256];
    const int tid = threadIdx.x;
    const int j   = blockIdx.x * 256 + tid;

    float pj[NOK];
#pragma unroll
    for (int t = 0; t < NOK; ++t) pj[t] = P[(size_t)j * NOK + t];
    float dj[4] = {0.f, 0.f, 0.f, 0.f};
    int cj = 0;
    if (CLS) cj = cls[j];
    else {
#pragma unroll
        for (int t = 0; t < 4; ++t) dj[t] = D[(size_t)j * 4 + t];
    }

    const int i0 = blockIdx.y * 256;
    for (int t = tid; t < 256 * NOK; t += 256) sP[t] = P[(size_t)i0 * NOK + t];
    if (CLS) sC[tid] = cls[i0 + tid];
    else {
#pragma unroll
        for (int t = 0; t < 4; ++t) sD[tid * 4 + t] = D[(size_t)(i0 + tid) * 4 + t];
    }
    __syncthreads();

    const float E2 = __expf(2.0f);
    float ex = 0.f, eg = 0.f;
    for (int i = 0; i < 256; ++i) {
        float x = 0.f;
#pragma unroll
        for (int t = 0; t < NOK; ++t) x += sP[i * NOK + t] * pj[t];
        ex += __expf(2.0f * x);
        if (CLS) {
            eg += (sC[i] == cj) ? E2 : 1.0f;
        } else {
            float gg = 0.f;
#pragma unroll
            for (int t = 0; t < 4; ++t) gg += sD[i * 4 + t] * dj[t];
            eg += __expf(2.0f * gg);
        }
    }
    atomicAdd(&colsum_x[j], ex);
    atomicAdd(&colsum_g[j], eg);
}

template <bool CLS>
__global__ __launch_bounds__(256)
void loss_pass2_kernel(const float* __restrict__ P,
                       const int* __restrict__ cls,
                       const float* __restrict__ D,
                       const float* __restrict__ colsum_x,
                       const float* __restrict__ colsum_g,
                       float* __restrict__ acc_out)
{
    __shared__ float sP[256 * NOK];
    __shared__ float sD[256 * 4];
    __shared__ int   sC[256];
    __shared__ float red[256];
    const int tid = threadIdx.x;
    const int j   = blockIdx.x * 256 + tid;

    float pj[NOK];
#pragma unroll
    for (int t = 0; t < NOK; ++t) pj[t] = P[(size_t)j * NOK + t];
    float dj[4] = {0.f, 0.f, 0.f, 0.f};
    int cj = 0;
    if (CLS) cj = cls[j];
    else {
#pragma unroll
        for (int t = 0; t < 4; ++t) dj[t] = D[(size_t)j * 4 + t];
    }
    const float denx  = colsum_x[j] + (float)N_ROWS * EPSF;
    const float deng  = colsum_g[j] + (float)N_ROWS * EPSF;
    const float ldx   = __logf(denx);
    const float invdg = 1.0f / deng;

    const int i0 = blockIdx.y * 256;
    for (int t = tid; t < 256 * NOK; t += 256) sP[t] = P[(size_t)i0 * NOK + t];
    if (CLS) sC[tid] = cls[i0 + tid];
    else {
#pragma unroll
        for (int t = 0; t < 4; ++t) sD[tid * 4 + t] = D[(size_t)(i0 + tid) * 4 + t];
    }
    __syncthreads();

    const float E2 = __expf(2.0f);
    float s = 0.f;
    for (int i = 0; i < 256; ++i) {
        float x = 0.f;
#pragma unroll
        for (int t = 0; t < NOK; ++t) x += sP[i * NOK + t] * pj[t];
        float w;
        if (CLS) {
            w = ((sC[i] == cj) ? E2 : 1.0f) * invdg;
        } else {
            float gg = 0.f;
#pragma unroll
            for (int t = 0; t < 4; ++t) gg += sD[i * 4 + t] * dj[t];
            w = __expf(2.0f * gg) * invdg;
        }
        s += (2.0f * x - ldx) * w;
    }
    red[tid] = s;
    __syncthreads();
    for (int st = 128; st > 0; st >>= 1) {
        if (tid < st) red[tid] += red[tid + st];
        __syncthreads();
    }
    if (tid == 0) atomicAdd(acc_out, red[0]);
}

// ---------------------------------------------------------------------------
// Output reductions
// ---------------------------------------------------------------------------
__global__ __launch_bounds__(256)
void col_mean8_kernel(const float* __restrict__ P, float* __restrict__ out8)
{
    __shared__ float red[256 * NOK];
    float a[NOK] = {0.f, 0.f, 0.f, 0.f, 0.f, 0.f, 0.f, 0.f};
    for (int m = threadIdx.x; m < N_ROWS; m += 256) {
#pragma unroll
        for (int j = 0; j < NOK; ++j) a[j] += P[(size_t)m * NOK + j];
    }
#pragma unroll
    for (int j = 0; j < NOK; ++j) red[threadIdx.x * NOK + j] = a[j];
    __syncthreads();
    for (int st = 128; st > 0; st >>= 1) {
        if (threadIdx.x < st) {
#pragma unroll
            for (int j = 0; j < NOK; ++j)
                red[threadIdx.x * NOK + j] += red[(threadIdx.x + st) * NOK + j];
        }
        __syncthreads();
    }
    if (threadIdx.x < NOK) out8[threadIdx.x] = red[threadIdx.x] * (1.0f / (float)N_ROWS);
}

__global__ __launch_bounds__(256)
void seg_accum_kernel(const float* __restrict__ P, const int* __restrict__ cls,
                      float* __restrict__ sums, float* __restrict__ counts)
{
    const int m = blockIdx.x * 256 + threadIdx.x;
    const int c = cls[m];
#pragma unroll
    for (int j = 0; j < NOK; ++j) atomicAdd(&sums[c * NOK + j], P[(size_t)m * NOK + j]);
    atomicAdd(&counts[c], 1.0f);
}

__global__ void seg_final_kernel(const float* __restrict__ sums,
                                 const float* __restrict__ counts,
                                 float* __restrict__ out)
{
    const int i = threadIdx.x;
    if (i < NUM_CLASSES * NOK) out[i] = sums[i] / fmaxf(counts[i >> 3], 1.0f);
}

__global__ void finalize_losses_kernel(const float* __restrict__ acc, float* __restrict__ out4)
{
    const int i = threadIdx.x;
    if (i < 4) out4[i] = -acc[i] * (1.0f / (float)N_ROWS);
}

__global__ void zero_kernel(float* __restrict__ p, int n)
{
    const int i = blockIdx.x * blockDim.x + threadIdx.x;
    if (i < n) p[i] = 0.0f;
}

// ---------------------------------------------------------------------------
extern "C" void kernel_launch(void* const* d_in, const int* in_sizes, int n_in,
                              void* d_out, int out_size, void* d_ws, size_t ws_size,
                              hipStream_t stream)
{
    (void)in_sizes; (void)n_in; (void)out_size; (void)ws_size;
    const float* rpn_feat    = (const float*)d_in[0];
    const float* rpn_deltas  = (const float*)d_in[1];
    const float* rpn_scale   = (const float*)d_in[2];
    const float* roi_feat    = (const float*)d_in[3];
    const float* roi_deltas  = (const float*)d_in[4];
    const float* roi_scale   = (const float*)d_in[5];
    const int*   rpn_class   = (const int*)d_in[6];
    const int*   roi_class   = (const int*)d_in[7];
    const float* W_rpn_feat  = (const float*)d_in[8];
    const float* b_rpn_feat  = (const float*)d_in[9];
    const float* W_rpn_del   = (const float*)d_in[10];
    const float* b_rpn_del   = (const float*)d_in[11];
    const float* W_rpn_sc    = (const float*)d_in[12];
    const float* b_rpn_sc    = (const float*)d_in[13];
    const float* W_roi_feat  = (const float*)d_in[14];
    const float* b_roi_feat  = (const float*)d_in[15];
    const float* W_roi_del   = (const float*)d_in[16];
    const float* b_roi_del   = (const float*)d_in[17];
    const float* W_roi_sc    = (const float*)d_in[18];
    const float* b_roi_sc    = (const float*)d_in[19];
    const float* rpn_cls_W1  = (const float*)d_in[20];
    const float* rpn_cls_b1  = (const float*)d_in[21];
    const float* rpn_cls_W2  = (const float*)d_in[22];
    const float* rpn_cls_b2  = (const float*)d_in[23];
    const float* rpn_bbox_W1 = (const float*)d_in[24];
    const float* rpn_bbox_b1 = (const float*)d_in[25];
    const float* rpn_bbox_W2 = (const float*)d_in[26];
    const float* rpn_bbox_b2 = (const float*)d_in[27];
    const float* roi_cls_W1  = (const float*)d_in[28];
    const float* roi_cls_b1  = (const float*)d_in[29];
    const float* roi_cls_W2  = (const float*)d_in[30];
    const float* roi_cls_b2  = (const float*)d_in[31];
    const int*   curr_iter   = (const int*)d_in[32];
    float* out = (float*)d_out;
    float* ws  = (float*)d_ws;

    // workspace layout (floats)
    float* sum_buf   = ws;                                  // 4096*2048
    float* h_buf     = sum_buf + (size_t)4096 * 2048;       // 4096*256
    float* alpha     = h_buf + (size_t)4096 * 256;          // 4 * 4096*8
    float* a_roi_cls  = alpha + 0 * (size_t)N_ROWS * NOK;
    float* a_rpn_cls  = alpha + 1 * (size_t)N_ROWS * NOK;
    float* a_roi_bbox = alpha + 2 * (size_t)N_ROWS * NOK;
    float* a_rpn_bbox = alpha + 3 * (size_t)N_ROWS * NOK;
    float* colx = alpha + 4 * (size_t)N_ROWS * NOK;         // 4*4096
    float* colg = colx + 4 * (size_t)N_ROWS;                // 4*4096
    float* acc  = colg + 4 * (size_t)N_ROWS;                // 4
    float* segs = acc + 4;                                  // 160
    float* segc = segs + NUM_CLASSES * NOK;                 // 20

    const int nz = 8 * N_ROWS + 4 + NUM_CLASSES * NOK + NUM_CLASSES;
    zero_kernel<<<(nz + 255) / 256, 256, 0, stream>>>(colx, nz);

    // --- RPN sum GEMM: [4096x256]@[512x256]^T + extras ---
    wmma_gemm_kernel<<<dim3(512 / 64, N_ROWS / 32), 32, 0, stream>>>(
        rpn_feat, 256, W_rpn_feat, sum_buf, 512, 256,
        b_rpn_feat, 2, rpn_deltas, W_rpn_del, b_rpn_del, rpn_scale, W_rpn_sc, b_rpn_sc);
    // rpn cls attention L1 (relu) + L2/softmax
    wmma_gemm_kernel<<<dim3(1, N_ROWS / 32), 32, 0, stream>>>(
        sum_buf, 512, rpn_cls_W1, h_buf, 64, 256,
        rpn_cls_b1, 1, nullptr, nullptr, nullptr, nullptr, nullptr, nullptr);
    attn_l2_softmax_kernel<<<16, 256, NOK * 64 * sizeof(float), stream>>>(
        h_buf, 64, rpn_cls_W2, rpn_cls_b2, curr_iter, a_rpn_cls);
    // rpn bbox attention
    wmma_gemm_kernel<<<dim3(1, N_ROWS / 32), 32, 0, stream>>>(
        sum_buf + 256, 512, rpn_bbox_W1, h_buf, 64, 256,
        rpn_bbox_b1, 1, nullptr, nullptr, nullptr, nullptr, nullptr, nullptr);
    attn_l2_softmax_kernel<<<16, 256, NOK * 64 * sizeof(float), stream>>>(
        h_buf, 64, rpn_bbox_W2, rpn_bbox_b2, curr_iter, a_rpn_bbox);

    // --- ROI sum GEMM: [4096x1024]@[2048x1024]^T + extras ---
    wmma_gemm_kernel<<<dim3(2048 / 64, N_ROWS / 32), 32, 0, stream>>>(
        roi_feat, 1024, W_roi_feat, sum_buf, 2048, 1024,
        b_roi_feat, 2, roi_deltas, W_roi_del, b_roi_del, roi_scale, W_roi_sc, b_roi_sc);
    // roi cls attention
    wmma_gemm_kernel<<<dim3(256 / 64, N_ROWS / 32), 32, 0, stream>>>(
        sum_buf, 2048, roi_cls_W1, h_buf, 256, 1024,
        roi_cls_b1, 1, nullptr, nullptr, nullptr, nullptr, nullptr, nullptr);
    attn_l2_softmax_kernel<<<16, 256, NOK * 256 * sizeof(float), stream>>>(
        h_buf, 256, roi_cls_W2, roi_cls_b2, curr_iter, a_roi_cls);
    // roi bbox attention (reference bug: reuses cls weights)
    wmma_gemm_kernel<<<dim3(256 / 64, N_ROWS / 32), 32, 0, stream>>>(
        sum_buf + 1024, 2048, roi_cls_W1, h_buf, 256, 1024,
        roi_cls_b1, 1, nullptr, nullptr, nullptr, nullptr, nullptr, nullptr);
    attn_l2_softmax_kernel<<<16, 256, NOK * 256 * sizeof(float), stream>>>(
        h_buf, 256, roi_cls_W2, roi_cls_b2, curr_iter, a_roi_bbox);

    // --- contrastive losses (acc order: roi_cls, rpn_cls, roi_bbox, rpn_bbox) ---
    dim3 lgrid(N_ROWS / 256, N_ROWS / 256);
    loss_pass1_kernel<true><<<lgrid, 256, 0, stream>>>(
        a_roi_cls, roi_class, nullptr, colx + 0 * N_ROWS, colg + 0 * N_ROWS);
    loss_pass2_kernel<true><<<lgrid, 256, 0, stream>>>(
        a_roi_cls, roi_class, nullptr, colx + 0 * N_ROWS, colg + 0 * N_ROWS, acc + 0);
    loss_pass1_kernel<true><<<lgrid, 256, 0, stream>>>(
        a_rpn_cls, rpn_class, nullptr, colx + 1 * N_ROWS, colg + 1 * N_ROWS);
    loss_pass2_kernel<true><<<lgrid, 256, 0, stream>>>(
        a_rpn_cls, rpn_class, nullptr, colx + 1 * N_ROWS, colg + 1 * N_ROWS, acc + 1);
    loss_pass1_kernel<false><<<lgrid, 256, 0, stream>>>(
        a_roi_bbox, nullptr, roi_deltas, colx + 2 * N_ROWS, colg + 2 * N_ROWS);
    loss_pass2_kernel<false><<<lgrid, 256, 0, stream>>>(
        a_roi_bbox, nullptr, roi_deltas, colx + 2 * N_ROWS, colg + 2 * N_ROWS, acc + 2);
    loss_pass1_kernel<false><<<lgrid, 256, 0, stream>>>(
        a_rpn_bbox, nullptr, rpn_deltas, colx + 3 * N_ROWS, colg + 3 * N_ROWS);
    loss_pass2_kernel<false><<<lgrid, 256, 0, stream>>>(
        a_rpn_bbox, nullptr, rpn_deltas, colx + 3 * N_ROWS, colg + 3 * N_ROWS, acc + 3);

    // --- outputs: rpn_cls_mean[8], rpn_bbox_mean[8], roi_cls_per_class[160],
    //              roi_bbox_mean[8], losses[4] ---
    col_mean8_kernel<<<1, 256, 0, stream>>>(a_rpn_cls, out + 0);
    col_mean8_kernel<<<1, 256, 0, stream>>>(a_rpn_bbox, out + 8);
    seg_accum_kernel<<<N_ROWS / 256, 256, 0, stream>>>(a_roi_cls, roi_class, segs, segc);
    seg_final_kernel<<<1, NUM_CLASSES * NOK, 0, stream>>>(segs, segc, out + 16);
    col_mean8_kernel<<<1, 256, 0, stream>>>(a_roi_bbox, out + 176);
    finalize_losses_kernel<<<1, 4, 0, stream>>>(acc, out + 184);
}
